// TemporalAttention_23733989278131
// MI455X (gfx1250) — compile-verified, matrix operand order checked by reading, not run on hardware
//
#include <hip/hip_runtime.h>
#include <hip/hip_bf16.h>
#include <hip/hip_fp16.h>

typedef __attribute__((ext_vector_type(16))) _Float16 v16h;
typedef __attribute__((ext_vector_type(8)))  _Float16 v8h;
typedef __attribute__((ext_vector_type(8)))  float    v8f;
typedef __attribute__((ext_vector_type(4)))  unsigned int u32x4;
typedef __attribute__((ext_vector_type(8)))  int      i32x8;
typedef __attribute__((ext_vector_type(4)))  int      i32x4;

#if __has_builtin(__builtin_amdgcn_tensor_load_to_lds)
#define USE_TDM 1
#else
#define USE_TDM 0
#endif

// ---------------------------------------------------------------------------
// WMMA helper: D = A(16x32 f16) * B(32x16 f16) + C(16x16 f32)
// ---------------------------------------------------------------------------
__device__ __forceinline__ v8f wmma_f16(v16h a, v16h b, v8f c) {
    return __builtin_amdgcn_wmma_f32_16x16x32_f16(
        false, a, false, b, (short)0, c, false, false);
}

// ---------------------------------------------------------------------------
// Fragment loader for 16x16x32 WMMA from a row-major f16 matrix.
//   lane L: row = L&15, K-chunks at kb=(L>>4)*8 and kb+16, 8 f16 each.
// Works for A (MxK row-major) and for B when stored as N x K row-major.
// ---------------------------------------------------------------------------
__device__ __forceinline__ v16h load_frag(const _Float16* __restrict__ p, int ld) {
    const int lane = threadIdx.x & 31;
    const int row  = lane & 15;
    const int kb   = (lane >> 4) << 3;
    const _Float16* q = p + (size_t)row * ld + kb;
    union { v16h v; v8h h[2]; } u;
    u.h[0] = *(const v8h*)(q);
    u.h[1] = *(const v8h*)(q + 16);
    return u.v;
}

// ---------------------------------------------------------------------------
// TDM: issue a 2D f16 tile load (tile_d0 contiguous elems x tile_d1 rows,
// row stride = stride0 elems) from global into LDS at lds_addr (bytes).
// D# layout per CDNA5 ISA ch.8: group0 = {count|flags, lds_addr, gaddr_lo,
// gaddr_hi|type}, group1 = dims/tile/strides. Groups 2/3 zeroed (2D tensor).
// ---------------------------------------------------------------------------
__device__ __forceinline__ void tdm_load_tile_f16(unsigned lds_addr,
                                                  const _Float16* gptr,
                                                  unsigned tensor_d0, unsigned tensor_d1,
                                                  unsigned stride0,
                                                  unsigned tile_d0, unsigned tile_d1) {
#if USE_TDM
    unsigned long long ga = (unsigned long long)(uintptr_t)gptr;
    u32x4 g0;
    g0.x = 1u;                                            // count=1, user descriptor
    g0.y = lds_addr;                                      // LDS byte address
    g0.z = (unsigned)ga;                                  // global_addr[31:0]
    g0.w = (unsigned)((ga >> 32) & 0x01ffffffu) | (2u << 30);  // addr[56:32] | type=2
    i32x8 g1;
    g1[0] = (int)(1u << 16);                              // data_size=1 -> 2 bytes
    g1[1] = (int)((tensor_d0 & 0xffffu) << 16);           // tensor_dim0[15:0]
    g1[2] = (int)((tensor_d0 >> 16) | ((tensor_d1 & 0xffffu) << 16));
    g1[3] = (int)((tensor_d1 >> 16) | (tile_d0 << 16));   // tensor_dim1 hi | tile_dim0
    g1[4] = (int)(tile_d1 & 0xffffu);                     // tile_dim1 ; tile_dim2=0
    g1[5] = (int)stride0;                                 // tensor_dim0_stride[31:0]
    g1[6] = 0;
    g1[7] = 0;
    i32x4 z4 = {0, 0, 0, 0};
#if __clang_major__ >= 23
    i32x8 z8 = {0, 0, 0, 0, 0, 0, 0, 0};
    __builtin_amdgcn_tensor_load_to_lds(g0, g1, z4, z4, z8, 0);
#else
    __builtin_amdgcn_tensor_load_to_lds(g0, g1, z4, z4, 0);
#endif
#endif
}

// ---------------------------------------------------------------------------
// Kernel 0: f32 -> f16 conversion (grid-stride)
// ---------------------------------------------------------------------------
__global__ __launch_bounds__(256) void cvt_f32_f16(const float* __restrict__ src,
                                                   _Float16* __restrict__ dst, int n) {
    int i = blockIdx.x * blockDim.x + threadIdx.x;
    int stride = gridDim.x * blockDim.x;
    for (; i < n; i += stride) dst[i] = (_Float16)src[i];
}

// ---------------------------------------------------------------------------
// Shared GEMM mainloop: C(128x128) += A(128xK) * B(128xK)^T, K step 32.
// 8 waves: wave = (wid&3)*32 rows x (wid>>2)*64 cols -> 2x4 accumulators,
// 8 WMMAs per K-step. A/B tiles staged to LDS by the TDM (double-buffered,
// prefetch tile k+1 while computing tile k) or by a fallback vector-copy.
// ---------------------------------------------------------------------------
__device__ __forceinline__ void gemm_tile_128(const _Float16* __restrict__ A,
                                              const _Float16* __restrict__ Bm,
                                              int K, int Mtot, int Ntot,
                                              int m0, int n0,
                                              _Float16* As, _Float16* Bs,   // [2][128*32]
                                              v8f c[2][4]) {
    const int tid = threadIdx.x;
    const int wid = tid >> 5;
    const int wm = (wid & 3) * 32;
    const int wn = (wid >> 2) * 64;
#if USE_TDM
    const bool issuer = (tid < 32);                  // wave 0 drives the TDM
    const unsigned ldsA = (unsigned)(uintptr_t)As;
    const unsigned ldsB = (unsigned)(uintptr_t)Bs;
    const unsigned bufBytes = 128u * 32u * 2u;       // 8 KB per buffer
    if (issuer) {
        tdm_load_tile_f16(ldsA, A + (size_t)m0 * K, K, Mtot, K, 32, 128);
        tdm_load_tile_f16(ldsB, Bm + (size_t)n0 * K, K, Ntot, K, 32, 128);
    }
#endif
    for (int k0 = 0, buf = 0; k0 < K; k0 += 32, buf ^= 1) {
#if USE_TDM
        if (issuer) {
            if (k0 + 32 < K) {                       // prefetch next tile, other buffer
                unsigned nb = (unsigned)(buf ^ 1) * bufBytes;
                tdm_load_tile_f16(ldsA + nb, A + (size_t)m0 * K + (k0 + 32), K, Mtot, K, 32, 128);
                tdm_load_tile_f16(ldsB + nb, Bm + (size_t)n0 * K + (k0 + 32), K, Ntot, K, 32, 128);
                __builtin_amdgcn_s_wait_tensorcnt(2);   // current tile (2 oldest ops) done
            } else {
                __builtin_amdgcn_s_wait_tensorcnt(0);
            }
        }
        __syncthreads();                              // LDS tile visible to all waves
#else
        for (int p = 0; p < 2; ++p) {                 // 512 chunks of 8 halves, 2 per thread
            int id = tid + p * 256;
            int lr = id >> 2, lc = (id & 3) * 8;
            *(v8h*)&As[buf * 4096 + lr * 32 + lc] = *(const v8h*)&A [(size_t)(m0 + lr) * K + k0 + lc];
            *(v8h*)&Bs[buf * 4096 + lr * 32 + lc] = *(const v8h*)&Bm[(size_t)(n0 + lr) * K + k0 + lc];
        }
        __syncthreads();
#endif
        const _Float16* as = As + buf * 4096;
        const _Float16* bs = Bs + buf * 4096;
        v16h a0 = load_frag(as + (wm +  0) * 32, 32);
        v16h a1 = load_frag(as + (wm + 16) * 32, 32);
        v16h b0 = load_frag(bs + (wn +  0) * 32, 32);
        v16h b1 = load_frag(bs + (wn + 16) * 32, 32);
        v16h b2 = load_frag(bs + (wn + 32) * 32, 32);
        v16h b3 = load_frag(bs + (wn + 48) * 32, 32);
        c[0][0] = wmma_f16(a0, b0, c[0][0]);
        c[0][1] = wmma_f16(a0, b1, c[0][1]);
        c[0][2] = wmma_f16(a0, b2, c[0][2]);
        c[0][3] = wmma_f16(a0, b3, c[0][3]);
        c[1][0] = wmma_f16(a1, b0, c[1][0]);
        c[1][1] = wmma_f16(a1, b1, c[1][1]);
        c[1][2] = wmma_f16(a1, b2, c[1][2]);
        c[1][3] = wmma_f16(a1, b3, c[1][3]);
        __syncthreads();                              // all reads done before next DMA/copy
    }
}

// ---------------------------------------------------------------------------
// Kernel 1: qkv = x @ w_qkv^T, scatter into Qh[b,h,t,d]*0.125, Kh[b,h,t,d],
//           Vt[b,h,d,t]. M=8192, N=3072, K=1024. Block tile 128x128.
// ---------------------------------------------------------------------------
__global__ __launch_bounds__(256) void gemm_qkv(const _Float16* __restrict__ A,   // [8192][1024]
                                                const _Float16* __restrict__ Bm,  // [3072][1024]
                                                _Float16* __restrict__ Qh,
                                                _Float16* __restrict__ Kh,
                                                _Float16* __restrict__ Vt) {
    __shared__ _Float16 As[2 * 128 * 32];
    __shared__ _Float16 Bs[2 * 128 * 32];
    const int tid  = threadIdx.x;
    const int wid  = tid >> 5;
    const int lane = tid & 31;
    const int m0 = blockIdx.y * 128;
    const int n0 = blockIdx.x * 128;
    const int wm = (wid & 3) * 32;
    const int wn = (wid >> 2) * 64;

    v8f c[2][4] = {{{}, {}, {}, {}}, {{}, {}, {}, {}}};
    gemm_tile_128(A, Bm, 1024, 8192, 3072, m0, n0, As, Bs, c);

    const int col   = lane & 15;
    const int rbase = (lane >> 4) * 8;
    for (int i = 0; i < 2; ++i)
        for (int jj = 0; jj < 4; ++jj) {
            int j   = n0 + wn + jj * 16 + col;      // 0..3071
            int sel = j >> 10;                      // 0=Q 1=K 2=V
            int cj  = j & 1023;
            int h   = cj >> 6;
            int d   = cj & 63;
            for (int r = 0; r < 8; ++r) {
                int row = m0 + wm + i * 16 + rbase + r;
                int b   = row >> 11;
                int tt  = row & 2047;
                float v = c[i][jj][r];
                size_t bh = (size_t)(b * 16 + h);
                if (sel == 0)      Qh[(bh * 2048 + tt) * 64 + d] = (_Float16)(v * 0.125f);
                else if (sel == 1) Kh[(bh * 2048 + tt) * 64 + d] = (_Float16)v;
                else               Vt[(bh * 64 + d) * 2048 + tt] = (_Float16)v;
            }
        }
}

// ---------------------------------------------------------------------------
// Kernel 2: causal flash attention, one wave per 16-query tile.
// ---------------------------------------------------------------------------
__global__ __launch_bounds__(256) void attn_flash(const _Float16* __restrict__ Qh,
                                                  const _Float16* __restrict__ Kh,
                                                  const _Float16* __restrict__ Vt,
                                                  _Float16* __restrict__ Y) {
    __shared__ _Float16 Pt[8][16 * 32];
    const int tid  = threadIdx.x;
    const int wid  = tid >> 5;
    const int lane = tid & 31;
    const int bh = blockIdx.x >> 4;
    const int qb = ((blockIdx.x & 15) * 8 + wid) * 16;
    const int b  = bh >> 4;
    const int h  = bh & 15;

    const _Float16* Qp = Qh + ((size_t)bh * 2048 + qb) * 64;
    const _Float16* Kp = Kh + (size_t)bh * 2048 * 64;
    const _Float16* Vp = Vt + (size_t)bh * 64 * 2048;      // [d][t], ld=2048

    v16h aq0 = load_frag(Qp, 64);
    v16h aq1 = load_frag(Qp + 32, 64);

    v8f acc[4] = {{}, {}, {}, {}};
    float mi[8], li[8];
    for (int r = 0; r < 8; ++r) { mi[r] = -1e30f; li[r] = 0.0f; }

    const int col   = lane & 15;
    const int rbase = (lane >> 4) * 8;

    for (int kb = 0; kb <= qb + 15; kb += 32) {
        __builtin_prefetch(Kp + (size_t)(kb + 32) * 64, 0, 0);   // next K tile
        v8f s0 = {}; v8f s1 = {};
        s0 = wmma_f16(aq0, load_frag(Kp + (size_t)kb * 64, 64), s0);
        s0 = wmma_f16(aq1, load_frag(Kp + (size_t)kb * 64 + 32, 64), s0);
        s1 = wmma_f16(aq0, load_frag(Kp + (size_t)(kb + 16) * 64, 64), s1);
        s1 = wmma_f16(aq1, load_frag(Kp + (size_t)(kb + 16) * 64 + 32, 64), s1);

        const int key0 = kb + col;
        const int key1 = key0 + 16;
        for (int r = 0; r < 8; ++r) {
            const int qrow = qb + rbase + r;
            float v0 = (key0 <= qrow) ? s0[r] : -1e30f;
            float v1 = (key1 <= qrow) ? s1[r] : -1e30f;
            float m = fmaxf(v0, v1);
            m = fmaxf(m, __shfl_xor(m, 1, 32));
            m = fmaxf(m, __shfl_xor(m, 2, 32));
            m = fmaxf(m, __shfl_xor(m, 4, 32));
            m = fmaxf(m, __shfl_xor(m, 8, 32));
            float mnew  = fmaxf(mi[r], m);
            float alpha = __expf(mi[r] - mnew);
            float p0 = __expf(v0 - mnew);
            float p1 = __expf(v1 - mnew);
            float rs = p0 + p1;
            rs += __shfl_xor(rs, 1, 32);
            rs += __shfl_xor(rs, 2, 32);
            rs += __shfl_xor(rs, 4, 32);
            rs += __shfl_xor(rs, 8, 32);
            li[r] = li[r] * alpha + rs;
            mi[r] = mnew;
            for (int t = 0; t < 4; ++t) acc[t][r] = acc[t][r] * alpha;
            const int mr = rbase + r;
            Pt[wid][mr * 32 + col]      = (_Float16)p0;
            Pt[wid][mr * 32 + col + 16] = (_Float16)p1;
        }
        asm volatile("s_wait_dscnt 0" ::: "memory");

        v16h ap = load_frag(&Pt[wid][0], 32);
        for (int t = 0; t < 4; ++t)
            acc[t] = wmma_f16(ap, load_frag(Vp + (size_t)(t * 16) * 2048 + kb, 2048), acc[t]);
    }

    _Float16* Yp = Y + ((size_t)(b * 2048 + qb)) * 1024 + h * 64;
    for (int t = 0; t < 4; ++t)
        for (int r = 0; r < 8; ++r) {
            float v = acc[t][r] / li[r];
            Yp[(size_t)(rbase + r) * 1024 + t * 16 + col] = (_Float16)v;
        }
}

// ---------------------------------------------------------------------------
// Kernel 3: out = y @ w_out^T  (M=8192, N=1024, K=1024), f32 output
// ---------------------------------------------------------------------------
__global__ __launch_bounds__(256) void gemm_out(const _Float16* __restrict__ A,   // [8192][1024]
                                                const _Float16* __restrict__ Bm,  // [1024][1024]
                                                float* __restrict__ out) {
    __shared__ _Float16 As[2 * 128 * 32];
    __shared__ _Float16 Bs[2 * 128 * 32];
    const int tid  = threadIdx.x;
    const int wid  = tid >> 5;
    const int lane = tid & 31;
    const int m0 = blockIdx.y * 128;
    const int n0 = blockIdx.x * 128;
    const int wm = (wid & 3) * 32;
    const int wn = (wid >> 2) * 64;

    v8f c[2][4] = {{{}, {}, {}, {}}, {{}, {}, {}, {}}};
    gemm_tile_128(A, Bm, 1024, 8192, 1024, m0, n0, As, Bs, c);

    const int col   = lane & 15;
    const int rbase = (lane >> 4) * 8;
    for (int i = 0; i < 2; ++i)
        for (int jj = 0; jj < 4; ++jj) {
            int j = n0 + wn + jj * 16 + col;
            for (int r = 0; r < 8; ++r) {
                int row = m0 + wm + i * 16 + rbase + r;
                out[(size_t)row * 1024 + j] = c[i][jj][r];
            }
        }
}

// ---------------------------------------------------------------------------
// Host launcher
// ---------------------------------------------------------------------------
extern "C" void kernel_launch(void* const* d_in, const int* in_sizes, int n_in,
                              void* d_out, int out_size, void* d_ws, size_t ws_size,
                              hipStream_t stream) {
    const float* x     = (const float*)d_in[0];   // [4,2048,1024]
    const float* w_qkv = (const float*)d_in[1];   // [3072,1024]
    const float* w_out = (const float*)d_in[2];   // [1024,1024]
    float* out = (float*)d_out;                   // [4,2048,1024]

    const int NX  = 4 * 2048 * 1024;              // 8388608
    const int NWQ = 3072 * 1024;                  // 3145728
    const int NWO = 1024 * 1024;                  // 1048576

    _Float16* ws = (_Float16*)d_ws;
    _Float16* Xh = ws;                 // 8388608 halves
    _Float16* Wq = Xh + NX;            // 3145728
    _Float16* Wo = Wq + NWQ;           // 1048576
    _Float16* Qh = Wo + NWO;           // 8388608 (pre-scaled by 1/8)
    _Float16* Kh = Qh + NX;            // 8388608
    _Float16* Vt = Kh + NX;            // 8388608 (transposed [b,h,d,t])
    _Float16* Y  = Vt + NX;            // 8388608

    cvt_f32_f16<<<2048, 256, 0, stream>>>(x, Xh, NX);
    cvt_f32_f16<<<2048, 256, 0, stream>>>(w_qkv, Wq, NWQ);
    cvt_f32_f16<<<1024, 256, 0, stream>>>(w_out, Wo, NWO);

    gemm_qkv<<<dim3(3072 / 128, 8192 / 128), 256, 0, stream>>>(Xh, Wq, Qh, Kh, Vt);
    attn_flash<<<dim3(64 * 16), 256, 0, stream>>>(Qh, Kh, Vt, Y);
    gemm_out<<<dim3(1024 / 128, 8192 / 128), 256, 0, stream>>>(Y, Wo, out);
}
